// MessagePassingNet_60722247631414
// MI455X (gfx1250) — compile-verified, matrix operand order checked by reading, not run on hardware
//
#include <hip/hip_runtime.h>
#include <hip/hip_bf16.h>
#include <stdint.h>

#define NA 100000   // atoms
#define NE 400000   // edges
#define NM 4000     // molecules

typedef _Float16 f16;
typedef __attribute__((ext_vector_type(16))) _Float16 v16h;
typedef __attribute__((ext_vector_type(8)))  _Float16 v8h;
typedef __attribute__((ext_vector_type(8)))  float    v8f;
typedef __attribute__((ext_vector_type(4)))  float    v4f;
typedef __attribute__((ext_vector_type(4)))  unsigned u32x4;
typedef __attribute__((ext_vector_type(8)))  int      i32x8;
typedef __attribute__((ext_vector_type(4)))  int      i32x4;

#if __has_builtin(__builtin_amdgcn_tensor_load_to_lds) && __has_builtin(__builtin_amdgcn_s_wait_tensorcnt)
#define HAVE_TDM 1
#else
#define HAVE_TDM 0
#endif

#define WMMA_F16(a, b, c) \
  __builtin_amdgcn_wmma_f32_16x16x32_f16(false, (a), false, (b), (short)0, (c), false, false)

__device__ __forceinline__ v8h ld8h(const f16* p) { return *(const v8h*)p; }

__device__ __forceinline__ v16h mk16(v8h a, v8h b) {
  v16h r;
#pragma unroll
  for (int i = 0; i < 8; ++i) { r[i] = a[i]; r[i + 8] = b[i]; }
  return r;
}

// load 8 consecutive f32, convert to 8 f16
__device__ __forceinline__ v8h cvt8(const float* p) {
  v4f a = *(const v4f*)p;
  v4f b = *(const v4f*)(p + 4);
  v8h r;
#pragma unroll
  for (int i = 0; i < 4; ++i) { r[i] = (f16)a[i]; r[i + 4] = (f16)b[i]; }
  return r;
}

// C/D fragment (lane: N = j*16+(lane&15); vgpr r: M = r + 8*(lane>>4)) -> bias+relu ->
// store into LDS in A-fragment order for the next GEMM (K = previous N).
// A-frag layout per ISA 7.12.2: lane m holds kl in {0..7,16..23}, lane m+16 holds {8..15,24..31}.
template <int NT>
__device__ __forceinline__ void bias_relu_repack(const v8f* acc, const float* bias,
                                                 f16* af, int lane) {
  const int m0 = lane & 15, hi = lane >> 4;
#pragma unroll
  for (int j = 0; j < NT; ++j) {
    float bv = bias[j * 16 + m0];
#pragma unroll
    for (int r = 0; r < 8; ++r) {
      float v = acc[j][r] + bv;
      v = v > 0.f ? v : 0.f;
      int m  = r + 8 * hi;
      int k  = j * 16 + m0;          // this output column becomes K next layer
      int kt = k >> 5, kl = k & 31;
      int dl = m + 16 * ((kl >> 3) & 1);
      int el = (kl & 7) + 8 * (kl >> 4);
      af[(kt * 32 + dl) * 16 + el] = (f16)v;
    }
  }
}

// ---- Pack a row-major [K,N] f32 weight into WMMA B-fragment order (f16). ----
// Fragment f = kt*NT + j; lane n0 holds column j*16+n0; element pattern mirrors A layout
// (B stored as "A of W^T"), so each lane's 16 halves are 32 contiguous bytes.
__global__ void pack_weight_kernel(const float* __restrict__ W, f16* __restrict__ dst,
                                   int K, int N) {
  int NT = N >> 4;
  int total = (K >> 5) * NT * 512;
  int idx = blockIdx.x * blockDim.x + threadIdx.x;
  if (idx >= total) return;
  int e    = idx & 15;
  int lane = (idx >> 4) & 31;
  int f    = idx >> 9;
  int j    = f % NT;
  int kt   = f / NT;
  int m0   = lane & 15, hi = lane >> 4;
  int n    = j * 16 + m0;
  int kl   = (e & 7) + 8 * hi + 16 * (e >> 3);
  int k    = kt * 32 + kl;
  dst[idx] = (f16)W[k * N + n];
}

// states f32 -> f16 copy + zero the next-step accumulator
__global__ void conv_zero_kernel(const float* __restrict__ src, f16* __restrict__ h,
                                 float* __restrict__ nxt, int n) {
  int i = blockIdx.x * blockDim.x + threadIdx.x;
  if (i < n) { h[i] = (f16)src[i]; nxt[i] = 0.f; }
}

__global__ void zero_kernel(float* __restrict__ p, int n) {
  int i = blockIdx.x * blockDim.x + threadIdx.x;
  if (i < n) p[i] = 0.f;
}

__global__ void mol_segsum_kernel(const float* __restrict__ st, const int* __restrict__ mid,
                                  float* __restrict__ molr) {
  int i = blockIdx.x * blockDim.x + threadIdx.x;
  if (i >= NA * 64) return;
  int a = i >> 6, f = i & 63;
  atomicAdd(&molr[(size_t)mid[a] * 64 + f], st[i]);
}

// ---- Edge MLP: one wave per 16-edge tile; 80 v_wmma per tile; atomic scatter-add. ----
__global__ __launch_bounds__(256) void edge_mlp_kernel(
    const f16* __restrict__ stH,
    const int* __restrict__ esrc, const int* __restrict__ edst,
    const f16* __restrict__ pWin, const f16* __restrict__ pWh, const f16* __restrict__ pWo,
    const float* __restrict__ bin, const float* __restrict__ bh, const float* __restrict__ bo,
    float* __restrict__ outStates) {
  __shared__ __attribute__((aligned(16))) f16 afrag[8][2048];  // 4KB A-frag buf per wave
  __shared__ __attribute__((aligned(16))) f16 woutL[8192];     // packed W_out, 16KB

#if HAVE_TDM
  // Stage packed W_out (16KB = 2048 x 8B) into LDS via the Tensor Data Mover.
  if (threadIdx.x == 0) {
    unsigned lds = (unsigned)(uintptr_t)(void*)&woutL[0];
    unsigned long long ga = (unsigned long long)(uintptr_t)pWo;
    // D# group0: count=1 | lds_addr | global_addr[56:0] | type=2 at [127:126]
    u32x4 g0 = { 1u, lds, (unsigned)ga,
                 ((unsigned)((ga >> 32) & 0x01FFFFFFull)) | 0x80000000u };
    // D# group1: data_size=3 (8B); tensor_dim0=2048; tile_dim0=2048; dim0_stride=2048; 1D
    i32x8 g1 = { (int)(3u << 16), (int)(2048u << 16), 0, (int)(2048u << 16),
                 0, 2048, 0, 0 };
    i32x4 z4 = { 0, 0, 0, 0 };
    i32x8 z8 = { 0, 0, 0, 0, 0, 0, 0, 0 };
    __builtin_amdgcn_tensor_load_to_lds(g0, g1, z4, z4, z8, 0);
    __builtin_amdgcn_s_wait_tensorcnt(0);
  }
#else
  for (int i = threadIdx.x; i < 8192; i += 256) woutL[i] = pWo[i];
#endif
  __syncthreads();

  const int wave = threadIdx.x >> 5;
  const int lane = threadIdx.x & 31;
  const int tile = blockIdx.x * 8 + wave;
  if (tile * 16 >= NE) return;
  const int eb = tile * 16;
  const int m0 = lane & 15, hi = lane >> 4;
  const int es = esrc[eb + m0];
  const int ed = edst[eb + m0];
  const f16* srow = stH + (size_t)es * 64;
  const f16* drow = stH + (size_t)ed * 64;

  v8f acc[8];
#pragma unroll
  for (int j = 0; j < 8; ++j)
#pragma unroll
    for (int r = 0; r < 8; ++r) acc[j][r] = 0.f;

  // GEMM1: concat(src,dst)[16x128] x W_in[128x128]
#pragma unroll
  for (int kt = 0; kt < 4; ++kt) {
    const f16* base = (kt < 2) ? srow : drow;
    int k0 = (kt & 1) * 32 + 8 * hi;                 // local K offset within 64-wide row
    v16h a = mk16(ld8h(base + k0), ld8h(base + k0 + 16));
#pragma unroll
    for (int j = 0; j < 8; ++j) {
      const f16* bp = pWin + ((kt * 8 + j) * 512 + lane * 16);
      v16h b = mk16(ld8h(bp), ld8h(bp + 8));
      acc[j] = WMMA_F16(a, b, acc[j]);
    }
  }
  bias_relu_repack<8>(acc, bin, afrag[wave], lane);
  asm volatile("s_wait_dscnt 0x0" ::: "memory");

  // GEMM2: H[16x128] x W_h[128x128]
#pragma unroll
  for (int j = 0; j < 8; ++j)
#pragma unroll
    for (int r = 0; r < 8; ++r) acc[j][r] = 0.f;
#pragma unroll
  for (int kt = 0; kt < 4; ++kt) {
    const f16* ap = &afrag[wave][(kt * 32 + lane) * 16];
    v16h a = mk16(ld8h(ap), ld8h(ap + 8));
#pragma unroll
    for (int j = 0; j < 8; ++j) {
      const f16* bp = pWh + ((kt * 8 + j) * 512 + lane * 16);
      v16h b = mk16(ld8h(bp), ld8h(bp + 8));
      acc[j] = WMMA_F16(a, b, acc[j]);
    }
  }
  bias_relu_repack<8>(acc, bh, afrag[wave], lane);
  asm volatile("s_wait_dscnt 0x0" ::: "memory");

  // GEMM3: H[16x128] x W_out[128x64]  (B from TDM-staged LDS)
#pragma unroll
  for (int j = 0; j < 4; ++j)
#pragma unroll
    for (int r = 0; r < 8; ++r) acc[j][r] = 0.f;
#pragma unroll
  for (int kt = 0; kt < 4; ++kt) {
    const f16* ap = &afrag[wave][(kt * 32 + lane) * 16];
    v16h a = mk16(ld8h(ap), ld8h(ap + 8));
#pragma unroll
    for (int j = 0; j < 4; ++j) {
      const f16* bp = &woutL[(kt * 4 + j) * 512 + lane * 16];
      v16h b = mk16(ld8h(bp), ld8h(bp + 8));
      acc[j] = WMMA_F16(a, b, acc[j]);
    }
  }

  // bias + relu + segment-sum scatter (f32 global atomics)
  int rows[8];
#pragma unroll
  for (int r = 0; r < 8; ++r) rows[r] = esrc[eb + r + 8 * hi];
#pragma unroll
  for (int j = 0; j < 4; ++j) {
    float bv = bo[j * 16 + m0];
#pragma unroll
    for (int r = 0; r < 8; ++r) {
      float v = acc[j][r] + bv;
      v = v > 0.f ? v : 0.f;
      atomicAdd(&outStates[(size_t)rows[r] * 64 + j * 16 + m0], v);
    }
  }
}

// ---- Readout MLP: one wave per 16-mol tile; 64->256->256->256 WMMA chain + final dot. ----
__global__ __launch_bounds__(128) void readout_kernel(
    const float* __restrict__ molr,
    const f16* __restrict__ pWin, const f16* __restrict__ pWh0, const f16* __restrict__ pWh1,
    const float* __restrict__ bin, const float* __restrict__ bh,
    const float* __restrict__ wout, const float* __restrict__ bout,
    float* __restrict__ out) {
  __shared__ __attribute__((aligned(16))) f16 afrag[4][4096];  // 8KB per wave (K up to 256)
  const int wave = threadIdx.x >> 5;
  const int lane = threadIdx.x & 31;
  const int tile = blockIdx.x * 4 + wave;
  if (tile * 16 >= NM) return;
  const int m0 = lane & 15, hi = lane >> 4;
  const float* row = molr + (size_t)(tile * 16 + m0) * 64;

  v8f acc[16];
#pragma unroll
  for (int j = 0; j < 16; ++j)
#pragma unroll
    for (int r = 0; r < 8; ++r) acc[j][r] = 0.f;

  // L1: mol_repr[16x64] x ro_W_in[64x256]
#pragma unroll
  for (int kt = 0; kt < 2; ++kt) {
    int k0 = kt * 32 + 8 * hi;
    v16h a = mk16(cvt8(row + k0), cvt8(row + k0 + 16));
#pragma unroll
    for (int j = 0; j < 16; ++j) {
      const f16* bp = pWin + ((kt * 16 + j) * 512 + lane * 16);
      v16h b = mk16(ld8h(bp), ld8h(bp + 8));
      acc[j] = WMMA_F16(a, b, acc[j]);
    }
  }
  bias_relu_repack<16>(acc, bin, afrag[wave], lane);
  asm volatile("s_wait_dscnt 0x0" ::: "memory");

  // L2 & L3: [16x256] x [256x256]
  const f16* Wl[2] = { pWh0, pWh1 };
  for (int l = 0; l < 2; ++l) {
#pragma unroll
    for (int j = 0; j < 16; ++j)
#pragma unroll
      for (int r = 0; r < 8; ++r) acc[j][r] = 0.f;
#pragma unroll
    for (int kt = 0; kt < 8; ++kt) {
      const f16* ap = &afrag[wave][(kt * 32 + lane) * 16];
      v16h a = mk16(ld8h(ap), ld8h(ap + 8));
#pragma unroll
      for (int j = 0; j < 16; ++j) {
        const f16* bp = Wl[l] + ((kt * 16 + j) * 512 + lane * 16);
        v16h b = mk16(ld8h(bp), ld8h(bp + 8));
        acc[j] = WMMA_F16(a, b, acc[j]);
      }
    }
    bias_relu_repack<16>(acc, bh + l * 256, afrag[wave], lane);
    asm volatile("s_wait_dscnt 0x0" ::: "memory");
  }

  // Final: out[m] = sum_k H3[m,k] * W_out[k] + b  (read H3 back from A-frag layout)
  if (lane < 16) {
    float s = bout[0];
    for (int k = 0; k < 256; ++k) {
      int kt = k >> 5, kl = k & 31;
      int dl = lane + 16 * ((kl >> 3) & 1);
      int el = (kl & 7) + 8 * (kl >> 4);
      s += (float)afrag[wave][(kt * 32 + dl) * 16 + el] * wout[k];
    }
    out[tile * 16 + lane] = s;
  }
}

extern "C" void kernel_launch(void* const* d_in, const int* in_sizes, int n_in,
                              void* d_out, int out_size, void* d_ws, size_t ws_size,
                              hipStream_t stream) {
  (void)in_sizes; (void)n_in; (void)out_size; (void)ws_size;
  const float* atom_states = (const float*)d_in[0];
  const int*   edge_src    = (const int*)d_in[1];
  const int*   edge_dst    = (const int*)d_in[2];
  const int*   mol_ids     = (const int*)d_in[3];
  const float* ms_W_in     = (const float*)d_in[4];
  const float* ms_b_in     = (const float*)d_in[5];
  const float* ms_W_h      = (const float*)d_in[6];
  const float* ms_b_h      = (const float*)d_in[7];
  const float* ms_W_out    = (const float*)d_in[8];
  const float* ms_b_out    = (const float*)d_in[9];
  const float* ro_W_in     = (const float*)d_in[10];
  const float* ro_b_in     = (const float*)d_in[11];
  const float* ro_W_h      = (const float*)d_in[12];
  const float* ro_b_h      = (const float*)d_in[13];
  const float* ro_W_out    = (const float*)d_in[14];
  const float* ro_b_out    = (const float*)d_in[15];

  char* ws = (char*)d_ws;
  size_t o = 0;
  float* stA = (float*)(ws + o); o += (size_t)NA * 64 * 4;
  float* stB = (float*)(ws + o); o += (size_t)NA * 64 * 4;
  f16*   stH = (f16*)(ws + o);   o += (size_t)NA * 64 * 2;
  f16* pWin[3]; f16* pWh[3]; f16* pWo[3];
  for (int s = 0; s < 3; ++s) {
    pWin[s] = (f16*)(ws + o); o += 32768;
    pWh[s]  = (f16*)(ws + o); o += 32768;
    pWo[s]  = (f16*)(ws + o); o += 16384;
  }
  f16* pRin = (f16*)(ws + o); o += 32768;
  f16* pRh0 = (f16*)(ws + o); o += 131072;
  f16* pRh1 = (f16*)(ws + o); o += 131072;
  float* molr = (float*)(ws + o); o += (size_t)NM * 64 * 4;

  auto pack = [&](const float* W, f16* dst, int K, int N) {
    int total = (K / 32) * (N / 16) * 512;
    pack_weight_kernel<<<(total + 255) / 256, 256, 0, stream>>>(W, dst, K, N);
  };
  for (int s = 0; s < 3; ++s) {
    pack(ms_W_in  + (size_t)s * 128 * 128, pWin[s], 128, 128);
    pack(ms_W_h   + (size_t)s * 128 * 128, pWh[s],  128, 128);
    pack(ms_W_out + (size_t)s * 128 * 64,  pWo[s],  128, 64);
  }
  pack(ro_W_in, pRin, 64, 256);
  pack(ro_W_h,             pRh0, 256, 256);
  pack(ro_W_h + 256 * 256, pRh1, 256, 256);

  const int nconv = NA * 64;
  const float* cur = atom_states;
  float* bufs[2] = { stA, stB };
  for (int s = 0; s < 3; ++s) {
    float* nxt = bufs[s & 1];
    conv_zero_kernel<<<(nconv + 255) / 256, 256, 0, stream>>>(cur, stH, nxt, nconv);
    edge_mlp_kernel<<<NE / 16 / 8, 256, 0, stream>>>(
        stH, edge_src, edge_dst, pWin[s], pWh[s], pWo[s],
        ms_b_in + (size_t)s * 128, ms_b_h + (size_t)s * 128, ms_b_out + (size_t)s * 64,
        nxt);
    cur = nxt;
  }

  zero_kernel<<<(NM * 64 + 255) / 256, 256, 0, stream>>>(molr, NM * 64);
  mol_segsum_kernel<<<(nconv + 255) / 256, 256, 0, stream>>>(cur, mol_ids, molr);

  readout_kernel<<<(NM / 16 + 3) / 4, 128, 0, stream>>>(
      molr, pRin, pRh0, pRh1, ro_b_in, ro_b_h, ro_W_out, ro_b_out, (float*)d_out);
}
